// feat_extraction_layer_67156108640285
// MI455X (gfx1250) — compile-verified, matrix operand order checked by reading, not run on hardware
//
#include <hip/hip_runtime.h>
#include <hip/hip_bf16.h>

// ---------------------------------------------------------------------------
// PointNet++ feature extraction on gfx1250 (MI455X), wave32 + WMMA.
//   - FPS: single 1024-thread block, register-resident distances, u64 argmax
//   - Ball query: one wave per centroid, ballot-ordered append
//   - Grouped MLP: v_wmma_f32_16x16x32_f16 GEMM, templated on tile counts so
//     all fragments/accumulators stay in statically-allocated VGPRs.
//     W transposed in LDS ([n][k], 96-half stride) -> B fragment = 2x b128 LDS
//     loads. BN+ReLU fused into the next layer's A-fragment load.
//     All addressing 32-bit (SGPR base + VGPR offset).
//   - Max-pool + final FC (WMMA, f32 out) -> d_out
// ---------------------------------------------------------------------------

typedef __attribute__((ext_vector_type(16))) _Float16 v16h;
typedef __attribute__((ext_vector_type(8)))  _Float16 v8h;
typedef __attribute__((ext_vector_type(8)))  float    v8f;

#define N0   12288     // raw points
#define NPT  10000     // NPOINT (FPS output count)

// ---------------------------------------------------------------- utilities
__device__ __forceinline__ unsigned long long u64_shfl_xor(unsigned long long v, int m) {
  int lo = __shfl_xor((int)(unsigned)v,          m, 32);
  int hi = __shfl_xor((int)(unsigned)(v >> 32),  m, 32);
  return ((unsigned long long)(unsigned)hi << 32) | (unsigned)lo;
}

// ------------------------------------------------- split pts -> xyz, normal
__global__ void k_split(const float* __restrict__ pts,
                        float* __restrict__ xyz, float* __restrict__ nrm) {
  int i = blockIdx.x * blockDim.x + threadIdx.x;
  if (i >= N0) return;
  for (int c = 0; c < 3; ++c) {
    xyz[i * 3 + c] = pts[c * N0 + i];
    nrm[i * 3 + c] = pts[(3 + c) * N0 + i];
  }
}

// ---------------------------------------------------- farthest point sample
// One block of 1024 threads (32 waves). Distances live in registers.
__global__ __launch_bounds__(1024)
void k_fps(const float* __restrict__ xyz, int N, int npoint,
           int* __restrict__ out_idx) {
  const int tid = threadIdx.x;
  const int T   = 1024;
  float dist[12];
#pragma unroll
  for (int k = 0; k < 12; ++k) {
    int i = tid + k * T;
    dist[k] = (i < N) ? 1e10f : -1.0f;
  }
  __shared__ int s_far;
  __shared__ unsigned long long s_w[32];
  if (tid == 0) s_far = 0;
  __syncthreads();

  const int lane = tid & 31;
  const int wv   = tid >> 5;

  for (int it = 0; it < npoint; ++it) {
    int far = s_far;
    if (tid == 0) out_idx[it] = far;
    float cx = xyz[far * 3 + 0], cy = xyz[far * 3 + 1], cz = xyz[far * 3 + 2];

    unsigned long long best = 0ull;
#pragma unroll
    for (int k = 0; k < 12; ++k) {
      int i = tid + k * T;
      if (i < N) {
        float dx = xyz[i * 3 + 0] - cx;
        float dy = xyz[i * 3 + 1] - cy;
        float dz = xyz[i * 3 + 2] - cz;
        float d  = dx * dx + dy * dy + dz * dz;
        float dd = fminf(dist[k], d);
        dist[k]  = dd;
        unsigned long long key =
            ((unsigned long long)__float_as_uint(dd) << 32) | (unsigned)i;
        best = (key > best) ? key : best;
      }
    }
#pragma unroll
    for (int off = 16; off > 0; off >>= 1) {
      unsigned long long o = u64_shfl_xor(best, off);
      best = (o > best) ? o : best;
    }
    if (lane == 0) s_w[wv] = best;
    __syncthreads();
    if (wv == 0) {
      unsigned long long b = s_w[lane];
#pragma unroll
      for (int off = 16; off > 0; off >>= 1) {
        unsigned long long o = u64_shfl_xor(b, off);
        b = (o > b) ? o : b;
      }
      if (lane == 0) s_far = (int)(b & 0xffffffffu);
    }
    __syncthreads();
  }
}

// ----------------------------------------------------------- gather 3-vecs
__global__ void k_gather3(const float* __restrict__ src,
                          const int* __restrict__ idx,
                          float* __restrict__ dst, int S) {
  int s = blockIdx.x * blockDim.x + threadIdx.x;
  if (s >= S) return;
  int p = idx[s];
  dst[s * 3 + 0] = src[p * 3 + 0];
  dst[s * 3 + 1] = src[p * 3 + 1];
  dst[s * 3 + 2] = src[p * 3 + 2];
}

// ------------------------------------------------------------- ball query
// One wave per centroid; ordered append of first `nsample` in-radius indices.
__global__ void k_ball(const float* __restrict__ xyz, int N,
                       const float* __restrict__ nxyz, int S,
                       float r2, int nsample, int* __restrict__ idx) {
  int w    = (blockIdx.x * blockDim.x + threadIdx.x) >> 5;
  int lane = threadIdx.x & 31;
  if (w >= S) return;
  float cx = nxyz[w * 3 + 0], cy = nxyz[w * 3 + 1], cz = nxyz[w * 3 + 2];
  int* row = idx + (size_t)w * nsample;
  int cnt = 0, first = -1;
  for (int base = 0; base < N; base += 32) {
    int i = base + lane;
    int ok = 0;
    if (i < N) {
      float dx = xyz[i * 3 + 0] - cx;
      float dy = xyz[i * 3 + 1] - cy;
      float dz = xyz[i * 3 + 2] - cz;
      ok = (dx * dx + dy * dy + dz * dz <= r2) ? 1 : 0;
    }
    unsigned m = (unsigned)__ballot(ok);
    if (ok) {
      int pos = cnt + __popc(m & ((1u << lane) - 1u));
      if (pos < nsample) row[pos] = i;
    }
    if (first < 0 && m) first = base + __ffs(m) - 1;
    cnt += __popc(m);
    if (cnt >= nsample) break;
  }
  if (first < 0) first = 0;
  int start = (cnt < nsample) ? cnt : nsample;
  for (int j = start + lane; j < nsample; j += 32) row[j] = first;
}

// ---------------------------------------------------------- WMMA MLP GEMM
// Y[M,Cout] = X[M,Cin] @ W + b, Cout = NT*16, Kp = KT*32 (compile-time).
// IM: 0 = grouped gather (xyz diff + feats), 1 = BN(scale/shift)+ReLU applied
//     to prev f16 activations (Cin == Kp), 2 = raw f32 input (Cin == Kp).
// OM: 0 = f16 out + BN-stat atomics, 1 = f32 out (no stats).
// All element offsets fit in 32 bits (max 1.28M * 64 = 82M elements).
template <int KT, int NT, int IM, int OM>
__global__ __launch_bounds__(256)
void k_mlp_wmma(int M, int Cin,
                const float* __restrict__ xyz,  const float* __restrict__ nxyz,
                const float* __restrict__ feat, const int* __restrict__ ball,
                int nsample,
                const _Float16* __restrict__ yprev,
                const float* __restrict__ xin,
                const float* __restrict__ ssg,   // [192]: scale[0..95], shift[96..191]
                const float* __restrict__ W, const float* __restrict__ bias,
                _Float16* __restrict__ yout, float* __restrict__ fout,
                float* __restrict__ sums)        // [2*Cout]: sum, sumsq
{
  constexpr int Kp   = KT * 32;
  constexpr int Cout = NT * 16;

  __shared__ __align__(16) _Float16 sWT[64 * 96];   // transposed: [n][k], stride 96
  __shared__ __align__(16) float    sSS[192];
  __shared__ float                  sStat[128];
  __shared__ float                  sB[64];

  const int tid = threadIdx.x;
  // load W (f32 -> f16), transposed, zero-padded in k
  for (int idx = tid; idx < Cout * Kp; idx += 256) {
    int n = idx / Kp, k = idx - n * Kp;
    float v = (k < Cin) ? W[k * Cout + n] : 0.f;
    sWT[n * 96 + k] = (_Float16)v;
  }
  if (tid < Cout) sB[tid] = bias[tid];
  if (IM == 1 && tid < Cin) { sSS[tid] = ssg[tid]; sSS[96 + tid] = ssg[96 + tid]; }
  if (tid < 128) sStat[tid] = 0.f;
  __syncthreads();

  const int lane = tid & 31;
  const int wv   = tid >> 5;
  const unsigned row0 = ((unsigned)blockIdx.x * 8 + wv) * 16;

  if (row0 < (unsigned)M) {
    const int      mloc = lane & 15;
    const unsigned mrow = row0 + mloc;
    const int      kgrp = (lane < 16) ? 0 : 8;

    unsigned ps = 0, pidx = 0;
    if (IM == 0) {
      ps   = mrow / (unsigned)nsample;
      pidx = (unsigned)ball[mrow];
    }

    v8f acc[NT];
#pragma unroll
    for (int nt = 0; nt < NT; ++nt)
#pragma unroll
      for (int r = 0; r < 8; ++r) acc[nt][r] = 0.f;

#pragma unroll
    for (int kt = 0; kt < KT; ++kt) {
      const int kbase = kt * 32 + kgrp;
      // ---- A fragment: 16x32 f16 (lanes 0-15: K kb+0..7, kb+16..23; 16-31: +8)
      v16h a;
      if (IM == 1) {
        const _Float16* yrow = yprev + mrow * (unsigned)Cin;
        v8h ylo = *(const v8h*)(yrow + kbase);
        v8h yhi = *(const v8h*)(yrow + kbase + 16);
#pragma unroll
        for (int h = 0; h < 8; ++h) {
          int k0 = kbase + h, k1 = kbase + 16 + h;
          a[h]     = (_Float16)fmaxf((float)ylo[h] * sSS[k0] + sSS[96 + k0], 0.f);
          a[8 + h] = (_Float16)fmaxf((float)yhi[h] * sSS[k1] + sSS[96 + k1], 0.f);
        }
      } else {
#pragma unroll
        for (int h = 0; h < 16; ++h) {
          int k = kbase + (h < 8 ? h : h + 8);
          float v = 0.f;
          if (IM == 0) {
            if (k < 3)        v = xyz[pidx * 3u + k] - nxyz[ps * 3u + k];
            else if (k < Cin) v = feat[pidx * (unsigned)(Cin - 3) + (k - 3)];
          } else {  // IM == 2
            v = xin[mrow * (unsigned)Cin + k];
          }
          a[h] = (_Float16)v;
        }
      }
      // ---- B fragments from transposed LDS W: two aligned b128 loads each
#pragma unroll
      for (int nt = 0; nt < NT; ++nt) {
        const int n = nt * 16 + mloc;
        v8h blo = *(const v8h*)(&sWT[n * 96 + kbase]);
        v8h bhi = *(const v8h*)(&sWT[n * 96 + kbase + 16]);
        v16h b = __builtin_shufflevector(blo, bhi,
                 0, 1, 2, 3, 4, 5, 6, 7, 8, 9, 10, 11, 12, 13, 14, 15);
        acc[nt] = __builtin_amdgcn_wmma_f32_16x16x32_f16(
            false, a, false, b, (short)0, acc[nt], false, false);
      }
    }

    const unsigned rbase = row0 + ((lane < 16) ? 0u : 8u);
#pragma unroll
    for (int nt = 0; nt < NT; ++nt) {
      const int n = nt * 16 + mloc;
      float bsum = 0.f, bsq = 0.f;
#pragma unroll
      for (int r = 0; r < 8; ++r) {
        unsigned off = (rbase + r) * (unsigned)Cout + n;
        float v = acc[nt][r] + sB[n];
        if (OM == 0) {
          yout[off] = (_Float16)v;
          bsum += v; bsq += v * v;
        } else {
          fout[off] = v;
        }
      }
      if (OM == 0) {
        atomicAdd(&sStat[n], bsum);        // ds_add_f32
        atomicAdd(&sStat[64 + n], bsq);
      }
    }
  }
  __syncthreads();
  if (OM == 0) {
    if (tid < Cout)                         atomicAdd(&sums[tid], sStat[tid]);
    else if (tid >= 64 && tid < 64 + Cout)  atomicAdd(&sums[Cout + (tid - 64)], sStat[tid]);
  }
}

// -------------------------------------------- BN stats -> scale/shift (192)
__global__ void k_stats(const float* __restrict__ sums, const float* __restrict__ g,
                        const float* __restrict__ be, int C, float invM,
                        float* __restrict__ ss) {
  int c = threadIdx.x;
  if (c >= C) return;
  float mean = sums[c] * invM;
  float var  = sums[C + c] * invM - mean * mean;
  float sc   = g[c] * rsqrtf(var + 1e-5f);
  ss[c]      = sc;
  ss[96 + c] = be[c] - mean * sc;
}

// --------------------------------------- max over samples of BN+ReLU output
__global__ void k_maxpool(const _Float16* __restrict__ y, int S, int ns, int C,
                          const float* __restrict__ ss, float* __restrict__ p) {
  int t = blockIdx.x * blockDim.x + threadIdx.x;
  if (t >= S * C) return;
  unsigned s = (unsigned)t / (unsigned)C, c = (unsigned)t % (unsigned)C;
  float sc = ss[c], sh = ss[96 + c];
  float m = 0.f;                         // ReLU => max >= 0
  const _Float16* row = y + s * (unsigned)(ns * C) + c;
  for (int j = 0; j < ns; ++j) {
    float v = (float)row[(unsigned)j * (unsigned)C];
    m = fmaxf(m, fmaxf(v * sc + sh, 0.f));
  }
  p[t] = m;
}

__global__ void k_copy(const float* __restrict__ src, float* __restrict__ dst, int n) {
  int i = blockIdx.x * blockDim.x + threadIdx.x;
  if (i < n) dst[i] = src[i];
}

// ===========================================================================
extern "C" void kernel_launch(void* const* d_in, const int* in_sizes, int n_in,
                              void* d_out, int out_size, void* d_ws, size_t ws_size,
                              hipStream_t stream) {
  (void)in_sizes; (void)n_in; (void)out_size; (void)ws_size;
  // input order: pts, sa1{W,b,g,be}x3, sa2{...}x2, sa3{...}x2, fc_W, fc_b
  const float* pts = (const float*)d_in[0];
  const float* W1[3]  = {(const float*)d_in[1], (const float*)d_in[5], (const float*)d_in[9]};
  const float* B1[3]  = {(const float*)d_in[2], (const float*)d_in[6], (const float*)d_in[10]};
  const float* G1[3]  = {(const float*)d_in[3], (const float*)d_in[7], (const float*)d_in[11]};
  const float* E1[3]  = {(const float*)d_in[4], (const float*)d_in[8], (const float*)d_in[12]};
  const float* W2[2]  = {(const float*)d_in[13], (const float*)d_in[17]};
  const float* B2[2]  = {(const float*)d_in[14], (const float*)d_in[18]};
  const float* G2[2]  = {(const float*)d_in[15], (const float*)d_in[19]};
  const float* E2[2]  = {(const float*)d_in[16], (const float*)d_in[20]};
  const float* W3[2]  = {(const float*)d_in[21], (const float*)d_in[25]};
  const float* B3[2]  = {(const float*)d_in[22], (const float*)d_in[26]};
  const float* G3[2]  = {(const float*)d_in[23], (const float*)d_in[27]};
  const float* E3[2]  = {(const float*)d_in[24], (const float*)d_in[28]};
  const float* fcW = (const float*)d_in[29];
  const float* fcB = (const float*)d_in[30];

  // workspace carve-up
  char* w = (char*)d_ws;
  auto alloc = [&](size_t bytes) -> void* {
    void* p = (void*)w; w += (bytes + 255) & ~(size_t)255; return p;
  };
  float*    xyz0 = (float*)alloc((size_t)N0 * 3 * 4);
  float*    nrm0 = (float*)alloc((size_t)N0 * 3 * 4);
  float*    nx1  = (float*)alloc((size_t)NPT * 3 * 4);
  float*    nx2  = (float*)alloc((size_t)NPT * 3 * 4);
  float*    nx3  = (float*)alloc((size_t)NPT * 3 * 4);
  float*    p1   = (float*)alloc((size_t)NPT * 64 * 4);
  float*    p2   = (float*)alloc((size_t)NPT * 64 * 4);
  float*    p3   = (float*)alloc((size_t)NPT * 32 * 4);
  int*      fidx = (int*)alloc((size_t)N0 * 4);
  int*      ball = (int*)alloc((size_t)NPT * 128 * 4);
  float*    sums = (float*)alloc(128 * 4);
  float*    ss   = (float*)alloc(192 * 4);
  _Float16* yA   = (_Float16*)alloc((size_t)NPT * 128 * 64 * 2);
  _Float16* yB   = (_Float16*)alloc((size_t)NPT * 128 * 64 * 2);

  float* out_xyz = (float*)d_out;            // 10000*3
  float* out_fc  = (float*)d_out + NPT * 3;  // 10000*32

  auto zstats = [&]() { (void)hipMemsetAsync(sums, 0, 128 * sizeof(float), stream); };
  auto nblk   = [](long long M) { return (int)((M / 16 + 7) / 8); };

  // ---------------- setup
  k_split<<<(N0 + 255) / 256, 256, 0, stream>>>(pts, xyz0, nrm0);

  // ---------------- SA1: N=12288 -> S=10000, ns=128, r=0.1, 6->32->32->64
  {
    const long long M = (long long)NPT * 128;
    k_fps<<<1, 1024, 0, stream>>>(xyz0, N0, NPT, fidx);
    k_gather3<<<(NPT + 255) / 256, 256, 0, stream>>>(xyz0, fidx, nx1, NPT);
    k_ball<<<(NPT * 32 + 255) / 256, 256, 0, stream>>>(xyz0, N0, nx1, NPT, 0.01f, 128, ball);
    zstats();
    k_mlp_wmma<1, 2, 0, 0><<<nblk(M), 256, 0, stream>>>((int)M, 6,
        xyz0, nx1, nrm0, ball, 128, nullptr, nullptr, ss, W1[0], B1[0], yA, nullptr, sums);
    k_stats<<<1, 64, 0, stream>>>(sums, G1[0], E1[0], 32, 1.f / (float)M, ss);
    zstats();
    k_mlp_wmma<1, 2, 1, 0><<<nblk(M), 256, 0, stream>>>((int)M, 32,
        nullptr, nullptr, nullptr, nullptr, 0, yA, nullptr, ss, W1[1], B1[1], yB, nullptr, sums);
    k_stats<<<1, 64, 0, stream>>>(sums, G1[1], E1[1], 32, 1.f / (float)M, ss);
    zstats();
    k_mlp_wmma<1, 4, 1, 0><<<nblk(M), 256, 0, stream>>>((int)M, 32,
        nullptr, nullptr, nullptr, nullptr, 0, yB, nullptr, ss, W1[2], B1[2], yA, nullptr, sums);
    k_stats<<<1, 64, 0, stream>>>(sums, G1[2], E1[2], 64, 1.f / (float)M, ss);
    k_maxpool<<<(NPT * 64 + 255) / 256, 256, 0, stream>>>(yA, NPT, 128, 64, ss, p1);
  }

  // ---------------- SA2: N=10000 -> S=10000, ns=64, r=0.2, 67->64->64
  {
    const long long M = (long long)NPT * 64;
    k_fps<<<1, 1024, 0, stream>>>(nx1, NPT, NPT, fidx);
    k_gather3<<<(NPT + 255) / 256, 256, 0, stream>>>(nx1, fidx, nx2, NPT);
    k_ball<<<(NPT * 32 + 255) / 256, 256, 0, stream>>>(nx1, NPT, nx2, NPT, 0.04f, 64, ball);
    zstats();
    k_mlp_wmma<3, 4, 0, 0><<<nblk(M), 256, 0, stream>>>((int)M, 67,
        nx1, nx2, p1, ball, 64, nullptr, nullptr, ss, W2[0], B2[0], yA, nullptr, sums);
    k_stats<<<1, 64, 0, stream>>>(sums, G2[0], E2[0], 64, 1.f / (float)M, ss);
    zstats();
    k_mlp_wmma<2, 4, 1, 0><<<nblk(M), 256, 0, stream>>>((int)M, 64,
        nullptr, nullptr, nullptr, nullptr, 0, yA, nullptr, ss, W2[1], B2[1], yB, nullptr, sums);
    k_stats<<<1, 64, 0, stream>>>(sums, G2[1], E2[1], 64, 1.f / (float)M, ss);
    k_maxpool<<<(NPT * 64 + 255) / 256, 256, 0, stream>>>(yB, NPT, 64, 64, ss, p2);
  }

  // ---------------- SA3: N=10000 -> S=10000, ns=32, r=0.4, 67->32->32
  {
    const long long M = (long long)NPT * 32;
    k_fps<<<1, 1024, 0, stream>>>(nx2, NPT, NPT, fidx);
    k_gather3<<<(NPT + 255) / 256, 256, 0, stream>>>(nx2, fidx, nx3, NPT);
    k_ball<<<(NPT * 32 + 255) / 256, 256, 0, stream>>>(nx2, NPT, nx3, NPT, 0.16f, 32, ball);
    zstats();
    k_mlp_wmma<3, 2, 0, 0><<<nblk(M), 256, 0, stream>>>((int)M, 67,
        nx2, nx3, p2, ball, 32, nullptr, nullptr, ss, W3[0], B3[0], yA, nullptr, sums);
    k_stats<<<1, 64, 0, stream>>>(sums, G3[0], E3[0], 32, 1.f / (float)M, ss);
    zstats();
    k_mlp_wmma<1, 2, 1, 0><<<nblk(M), 256, 0, stream>>>((int)M, 32,
        nullptr, nullptr, nullptr, nullptr, 0, yA, nullptr, ss, W3[1], B3[1], yB, nullptr, sums);
    k_stats<<<1, 64, 0, stream>>>(sums, G3[1], E3[1], 32, 1.f / (float)M, ss);
    k_maxpool<<<(NPT * 32 + 255) / 256, 256, 0, stream>>>(yB, NPT, 32, 32, ss, p3);
  }

  // ---------------- outputs: new_xyz3 and p3 @ fc_W + fc_b
  k_copy<<<(NPT * 3 + 255) / 256, 256, 0, stream>>>(nx3, out_xyz, NPT * 3);
  k_mlp_wmma<1, 2, 2, 1><<<nblk((long long)NPT), 256, 0, stream>>>(NPT, 32,
      nullptr, nullptr, nullptr, nullptr, 0, nullptr, p3, ss, fcW, fcB, nullptr, out_fc, sums);
}
